// SSM_31293131718897
// MI455X (gfx1250) — matrix operand order repaired and checked by reading, not compiled, not run on hardware
//
#include <hip/hip_runtime.h>
#include <stdint.h>

#define K_TOP 32
#define B_    2
#define C_    1024
#define H_    48
#define W_    48
#define HW    2304
#define IMG_  768

typedef __attribute__((ext_vector_type(16))) __bf16        v16bf;
typedef __attribute__((ext_vector_type(8)))  float         v8f;
typedef __attribute__((ext_vector_type(4)))  unsigned int  u32x4;

__device__ __forceinline__ unsigned short f2bf(float f) {
  unsigned int u = __float_as_uint(f);
  u += 0x7FFFu + ((u >> 16) & 1u);       // round-to-nearest-even
  return (unsigned short)(u >> 16);
}

// ---------------------------------------------------------------------------
// 1) per-pixel L2 normalize over channels, write bf16 normalized features
// ---------------------------------------------------------------------------
__global__ __launch_bounds__(256) void k_normalize(
    const float* __restrict__ ref, const float* __restrict__ cur,
    unsigned short* __restrict__ refn, unsigned short* __restrict__ curn) {
  const int p = blockIdx.x * 256 + threadIdx.x;          // [0, B*HW)
  const float* src = blockIdx.y ? cur : ref;
  unsigned short* dst = blockIdx.y ? curn : refn;
  const int b = p / HW, pl = p % HW;
  const size_t base = (size_t)b * C_ * HW + pl;
  float ss = 0.f;
  for (int c = 0; c < C_; ++c) { float v = src[base + (size_t)c * HW]; ss += v * v; }
  const float inv = 1.f / fmaxf(sqrtf(ss), 1e-12f);
  for (int c = 0; c < C_; ++c)
    dst[base + (size_t)c * HW] = f2bf(src[base + (size_t)c * HW] * inv);
}

// ---------------------------------------------------------------------------
// 2) bilinear 768 -> 48 downsample (half-pixel centers -> exact 4-tap avg)
// ---------------------------------------------------------------------------
__global__ __launch_bounds__(256) void k_mask(
    const float* __restrict__ m, float* __restrict__ md) {
  const int p = blockIdx.x * 256 + threadIdx.x;          // [0, B*HW)
  const int b = p / HW, pl = p % HW;
  const int y = pl / W_, x = pl % W_;
  const int sy = 16 * y + 7, sx = 16 * x + 7;
  const float* mb = m + (size_t)b * IMG_ * IMG_;
  md[p] = 0.25f * (mb[(size_t)sy * IMG_ + sx]       + mb[(size_t)sy * IMG_ + sx + 1] +
                   mb[(size_t)(sy + 1) * IMG_ + sx] + mb[(size_t)(sy + 1) * IMG_ + sx + 1]);
}

// ---------------------------------------------------------------------------
// 3) corr[b,i,j] = sum_c curn[b,c,i]*refn[b,c,j]  -- WMMA bf16, f32 accumulate
//    128 threads = 4 waves; block tile 64x64; wave tile 32x32 (4 WMMA/k-step,
//    2x fragment reuse); K-step 32; LDS-staged transposed, 80B padded rows
// ---------------------------------------------------------------------------
__global__ __launch_bounds__(128) void k_corr(
    const unsigned short* __restrict__ curn,
    const unsigned short* __restrict__ refn,
    float* __restrict__ corr) {
  // [buf][row m|n (64)][k (32, padded to 40)]: 16B-aligned rows, bank-clean
  __shared__ __align__(16) unsigned short lds[2][64][40];
  const int t    = threadIdx.x;
  const int b    = blockIdx.z;
  const int i0   = blockIdx.y * 64;
  const int j0   = blockIdx.x * 64;
  const int wid  = t >> 5, lane = t & 31;
  const int mi   = wid >> 1, ni = wid & 1;               // 2x2 wave subtiles of 32x32
  const int half = lane >> 4, r0 = lane & 15;
  const int fk   = t & 31,  fm = (t >> 5) * 16;          // fill: 1 k-col, 16 m-rows

  union U8   { u32x4 q;    unsigned short s[8]; };
  union Frag { u32x4 q[2]; v16bf v; };
  v8f acc00 = {}, acc01 = {}, acc10 = {}, acc11 = {};

  const size_t rowA = (size_t)(b * C_) * HW + i0 + fm;
  const size_t rowB = (size_t)(b * C_) * HW + j0 + fm;

  for (int c0 = 0; c0 < C_; c0 += 32) {
    __syncthreads();
    const size_t go = (size_t)(c0 + fk) * HW;
    U8 ta0, ta1, tb0, tb1;
    ta0.q = *(const u32x4*)(curn + rowA + go);
    ta1.q = *(const u32x4*)(curn + rowA + go + 8);
    tb0.q = *(const u32x4*)(refn + rowB + go);
    tb1.q = *(const u32x4*)(refn + rowB + go + 8);
#pragma unroll
    for (int r = 0; r < 8; ++r) {
      lds[0][fm + r][fk]     = ta0.s[r];                 // lds_a[m][k] = curn[k][i0+m]
      lds[0][fm + 8 + r][fk] = ta1.s[r];
      lds[1][fm + r][fk]     = tb0.s[r];                 // lds_b[n][k] = refn[k][j0+n]
      lds[1][fm + 8 + r][fk] = tb1.s[r];
    }
    __syncthreads();
    Frag fa0, fa1, fb0, fb1;
    const int k0  = 8 * half;                            // A: K in [8h,8h+8) U [16+8h,+8)
    const int kb  = 8 * half;                            // B: K in [8h,8h+8) U [16+8h,+8)
    const int ra0 = mi * 32 + r0,      ra1 = mi * 32 + 16 + r0;
    const int rb0 = ni * 32 + r0,      rb1 = ni * 32 + 16 + r0;
    fa0.q[0] = *(const u32x4*)&lds[0][ra0][k0];
    fa0.q[1] = *(const u32x4*)&lds[0][ra0][16 + k0];
    fa1.q[0] = *(const u32x4*)&lds[0][ra1][k0];
    fa1.q[1] = *(const u32x4*)&lds[0][ra1][16 + k0];
    fb0.q[0] = *(const u32x4*)&lds[1][rb0][kb];
    fb0.q[1] = *(const u32x4*)&lds[1][rb0][kb + 16];
    fb1.q[0] = *(const u32x4*)&lds[1][rb1][kb];
    fb1.q[1] = *(const u32x4*)&lds[1][rb1][kb + 16];
    acc00 = __builtin_amdgcn_wmma_f32_16x16x32_bf16(false, fa0.v, false, fb0.v, (short)0, acc00, false, false);
    acc01 = __builtin_amdgcn_wmma_f32_16x16x32_bf16(false, fa0.v, false, fb1.v, (short)0, acc01, false, false);
    acc10 = __builtin_amdgcn_wmma_f32_16x16x32_bf16(false, fa1.v, false, fb0.v, (short)0, acc10, false, false);
    acc11 = __builtin_amdgcn_wmma_f32_16x16x32_bf16(false, fa1.v, false, fb1.v, (short)0, acc11, false, false);
  }
  // C/D layout: VGPR r -> M = r + 8*half, N = lane&15
#pragma unroll
  for (int si = 0; si < 2; ++si) {
#pragma unroll
    for (int sj = 0; sj < 2; ++sj) {
      v8f a = si ? (sj ? acc11 : acc10) : (sj ? acc01 : acc00);
      size_t ob = ((size_t)b * HW + i0 + mi * 32 + si * 16 + 8 * half) * HW
                + j0 + ni * 32 + sj * 16 + r0;
#pragma unroll
      for (int r = 0; r < 8; ++r) corr[ob + (size_t)r * HW] = a[r];
    }
  }
}

// ---------------------------------------------------------------------------
// 4) masked top-32 per (b, cur pixel, fg/bg); fg pass also writes row score
// ---------------------------------------------------------------------------
__global__ __launch_bounds__(32) void k_topk(
    const float* __restrict__ corr, const float* __restrict__ md,
    float* __restrict__ out, float* __restrict__ score) {
  const int i = blockIdx.x, fg = blockIdx.y, b = blockIdx.z, lane = threadIdx.x;
  const float* row  = corr + ((size_t)b * HW + i) * HW;
  const float* mrow = md + (size_t)b * HW;
  float vals[72];
#pragma unroll
  for (int t = 0; t < 72; ++t) {
    const int j = t * 32 + lane;
    const bool s = fg ? (mrow[j] > 0.5f) : (mrow[j] <= 0.5f);
    vals[t] = s ? row[j] : 0.0f;                         // masked-out -> 0 (matches corr*sel)
  }
  if (fg) {
    float s = 0.f;
#pragma unroll
    for (int t = 0; t < 72; ++t) s += vals[t];
#pragma unroll
    for (int off = 16; off >= 1; off >>= 1) s += __shfl_xor(s, off, 32);
    if (lane == 0) score[(size_t)b * HW + i] = s;
  }
  const int chbase = b * 128 + (fg ? 32 : 0);            // [bg | fg] channel order
  for (int r = 0; r < K_TOP; ++r) {
    float mv = -3.0e38f;
#pragma unroll
    for (int t = 0; t < 72; ++t) mv = fmaxf(mv, vals[t]);
    float wm = mv;
#pragma unroll
    for (int off = 16; off >= 1; off >>= 1) wm = fmaxf(wm, __shfl_xor(wm, off, 32));
    const unsigned long long bal = __ballot(mv == wm);
    const int chosen = (int)(__ffsll((long long)bal) - 1);
    if (lane == chosen) {
      bool done = false;
#pragma unroll
      for (int t = 0; t < 72; ++t)
        if (!done && vals[t] == wm) { vals[t] = -3.0e38f; done = true; }
    }
    if (lane == 0) out[(size_t)(chbase + r) * HW + i] = wm;
  }
}

// ---------------------------------------------------------------------------
// 5) top-32 indices of score per batch (one wave per batch)
// ---------------------------------------------------------------------------
__global__ __launch_bounds__(32) void k_scoretop(
    const float* __restrict__ score, int* __restrict__ sidx) {
  const int b = blockIdx.x, lane = threadIdx.x;
  float vals[72];
#pragma unroll
  for (int t = 0; t < 72; ++t) vals[t] = score[(size_t)b * HW + t * 32 + lane];
  for (int r = 0; r < K_TOP; ++r) {
    float mv = -3.0e38f;
#pragma unroll
    for (int t = 0; t < 72; ++t) mv = fmaxf(mv, vals[t]);
    float wm = mv;
#pragma unroll
    for (int off = 16; off >= 1; off >>= 1) wm = fmaxf(wm, __shfl_xor(wm, off, 32));
    const unsigned long long bal = __ballot(mv == wm);
    const int chosen = (int)(__ffsll((long long)bal) - 1);
    if (lane == chosen) {
      bool done = false;
#pragma unroll
      for (int t = 0; t < 72; ++t)
        if (!done && vals[t] == wm) {
          vals[t] = -3.0e38f; sidx[b * K_TOP + r] = t * 32 + lane; done = true;
        }
    }
  }
}

// ---------------------------------------------------------------------------
// 6) struct[b,kk,p] = sum_c ref[b,c,idx[b,kk]] * ref[b,c,p]
// ---------------------------------------------------------------------------
__global__ __launch_bounds__(256) void k_struct(
    const float* __restrict__ ref, const int* __restrict__ sidx,
    float* __restrict__ sw) {
  const int p = blockIdx.x * 256 + threadIdx.x;
  const int kk = blockIdx.y, b = blockIdx.z;
  const int ridx = sidx[b * K_TOP + kk];
  const float* rf = ref + (size_t)b * C_ * HW;
  float acc = 0.f;
  for (int c = 0; c < C_; ++c)
    acc += rf[(size_t)c * HW + ridx] * rf[(size_t)c * HW + p];
  sw[((size_t)b * K_TOP + kk) * HW + p] = acc;
}

// ---------------------------------------------------------------------------
// 7) grouped conv + global conv + struct fusion + final 2-ch projection
// ---------------------------------------------------------------------------
__global__ __launch_bounds__(64) void k_combine(
    const float* __restrict__ ref, const float* __restrict__ sw,
    const float* __restrict__ gw,  const float* __restrict__ gb,
    const float* __restrict__ glw, const float* __restrict__ glb,
    const float* __restrict__ cw,  const float* __restrict__ cb,
    float* __restrict__ out) {
  const int p = blockIdx.x * 64 + threadIdx.x;           // [0, B*HW)
  const int b = p / HW, pl = p % HW;
  const float* rf = ref + (size_t)b * C_ * HW;

  float sval[32]; float ssum = 0.f;
#pragma unroll
  for (int g = 0; g < 32; ++g) { sval[g] = sw[((size_t)b * 32 + g) * HW + pl]; ssum += sval[g]; }

  float gs[32];
#pragma unroll
  for (int o = 0; o < 32; ++o) gs[o] = 0.f;
  for (int g = 0; g < 32; ++g) {                         // grouped 1x1 conv + relu
    float rin[32];
#pragma unroll
    for (int i = 0; i < 32; ++i) rin[i] = rf[(size_t)(g * 32 + i) * HW + pl];
    const float sg = sval[g];
    for (int o = 0; o < 32; ++o) {
      float a = gb[g * 32 + o];
#pragma unroll
      for (int i = 0; i < 32; ++i) a += rin[i] * gw[(g * 32 + o) * 32 + i];
      gs[o] += sg * fmaxf(a, 0.f);                       // sum over groups
    }
  }

  float ga[32];
#pragma unroll
  for (int o = 0; o < 32; ++o) ga[o] = 0.f;
  for (int c = 0; c < C_; ++c) {                         // global 1x1 conv
    const float x = rf[(size_t)c * HW + pl];
#pragma unroll
    for (int o = 0; o < 32; ++o) ga[o] += x * glw[o * C_ + c];
  }
  const float smean = ssum * (1.0f / 32.0f);
  float glbv[32];
#pragma unroll
  for (int o = 0; o < 32; ++o) glbv[o] = smean * fmaxf(ga[o] + glb[o], 0.f);

#pragma unroll
  for (int o = 0; o < 32; ++o) {
    out[((size_t)b * 128 + 64 + o) * HW + pl] = gs[o];   // group_struct
    out[((size_t)b * 128 + 96 + o) * HW + pl] = glbv[o]; // global_struct
  }
  float* out2 = out + (size_t)B_ * 128 * HW;             // struct_mask
#pragma unroll
  for (int o2 = 0; o2 < 2; ++o2) {
    float a = cb[o2];
#pragma unroll
    for (int o = 0; o < 32; ++o)
      a += gs[o] * cw[o2 * 64 + o] + glbv[o] * cw[o2 * 64 + 32 + o];
    out2[((size_t)b * 2 + o2) * HW + pl] = a;
  }
}

// ---------------------------------------------------------------------------
extern "C" void kernel_launch(void* const* d_in, const int* in_sizes, int n_in,
                              void* d_out, int out_size, void* d_ws, size_t ws_size,
                              hipStream_t stream) {
  (void)in_sizes; (void)n_in; (void)out_size; (void)ws_size;
  const float* ref = (const float*)d_in[0];
  const float* cur = (const float*)d_in[1];
  const float* msk = (const float*)d_in[2];
  const float* gw  = (const float*)d_in[3];
  const float* gb  = (const float*)d_in[4];
  const float* glw = (const float*)d_in[5];
  const float* glb = (const float*)d_in[6];
  const float* cw  = (const float*)d_in[7];
  const float* cb  = (const float*)d_in[8];
  float* out = (float*)d_out;

  char* ws = (char*)d_ws;                                // ~62 MB scratch layout
  unsigned short* refn = (unsigned short*)(ws + (size_t)0);          // 9,437,184 B
  unsigned short* curn = (unsigned short*)(ws + (size_t)9437184);    // 9,437,184 B
  float* corr  = (float*)(ws + (size_t)18874368);                    // 42,467,328 B
  float* md    = (float*)(ws + (size_t)61341696);                    // 18,432 B
  float* score = (float*)(ws + (size_t)61360128);                    // 18,432 B
  int*   sidx  = (int*)  (ws + (size_t)61378560);                    // 256 B
  float* swb   = (float*)(ws + (size_t)61378816);                    // 589,824 B

  k_normalize<<<dim3(18, 2), 256, 0, stream>>>(ref, cur, refn, curn);
  k_mask     <<<18, 256, 0, stream>>>(msk, md);
  k_corr     <<<dim3(36, 36, 2), 128, 0, stream>>>(curn, refn, corr);
  k_topk     <<<dim3(HW, 2, 2), 32, 0, stream>>>(corr, md, out, score);
  k_scoretop <<<2, 32, 0, stream>>>(score, sidx);
  k_struct   <<<dim3(9, 32, 2), 256, 0, stream>>>(ref, sidx, swb);
  k_combine  <<<72, 64, 0, stream>>>(ref, swb, gw, gb, glw, glb, cw, cb, out);
}
